// GAT_69106023793064
// MI455X (gfx1250) — compile-verified, matrix operand order checked by reading, not run on hardware
//
#include <hip/hip_runtime.h>
#include <hip/hip_bf16.h>

// ---------------- problem constants (match reference) ----------------
#define NN     50000
#define MPAD   50016       // NN padded to multiple of 32 for 2x2 WMMA tiling
#define E0C    800000
#define ETOT   850000      // E0 + N self loops
#define INDIM  256
#define HIDDIM 256
#define OUTDIM 64
#define H1C    8
#define C1C    32
#define H2C    1
#define C2C    64
#define NEG_SLOPE 0.2f

typedef __attribute__((ext_vector_type(16))) _Float16 v16h;
typedef __attribute__((ext_vector_type(8)))  _Float16 v8h;
typedef __attribute__((ext_vector_type(8)))  float    v8f;

union V16U { v16h v; v8h h[2]; };

// monotone float<->uint encoding for atomic max on floats (init 0 == -inf)
__device__ __forceinline__ unsigned fenc(float f) {
    int b = __float_as_int(f);
    return (b >= 0) ? ((unsigned)b | 0x80000000u) : ~(unsigned)b;
}
__device__ __forceinline__ float fdec(unsigned u) {
    int b = (u & 0x80000000u) ? (int)(u & 0x7FFFFFFFu) : (int)(~u);
    return __int_as_float(b);
}

// ---------------- prep kernels ----------------
__global__ void cvt_f32_to_f16(const float* __restrict__ src,
                               _Float16* __restrict__ dst, int n) {
    int i = blockIdx.x * blockDim.x + threadIdx.x;
    if (i < n) dst[i] = (_Float16)src[i];
}

// W [K, Nc] row-major f32  ->  Wt [Nc, K] row-major f16
__global__ void xpose_cvt_f16(const float* __restrict__ W,
                              _Float16* __restrict__ Wt, int K, int Nc) {
    int i = blockIdx.x * blockDim.x + threadIdx.x;
    if (i >= K * Nc) return;
    int k = i / Nc, n = i % Nc;
    Wt[(size_t)n * K + k] = (_Float16)W[i];
}

// ---------------- WMMA GEMM: C[MPAD,Nc] = A[MPAD,K] * Bt[Nc,K]^T ----------
// 2x2 register blocking: one wave computes a 32x32 output tile with four
// 16x16 f32 accumulators; each A/B fragment is reused twice -> 2 b128 loads
// per v_wmma_f32_16x16x32_f16 in the K-loop.
// 16-bit A fragment (16x32): lanes 0-15 hold row r, K {k0..k0+7, k0+16..k0+23};
// lanes 16-31 hold row r, K {k0+8..k0+15, k0+24..k0+31}.  B mirrored on cols.
#define WMMA16(a, b, c) __builtin_amdgcn_wmma_f32_16x16x32_f16( \
        false, (a), false, (b), (short)0, (c), false, false)

__global__ __launch_bounds__(128)
void gemm_f16_wmma(const _Float16* __restrict__ A,
                   const _Float16* __restrict__ Bt,
                   float* __restrict__ C, int Mtiles2, int Nc, int K) {
    int wave = (int)((blockIdx.x * blockDim.x + threadIdx.x) >> 5);
    int lane = threadIdx.x & 31;
    int nt2 = Nc >> 5;                 // 32-wide N tiles
    int tm = wave / nt2;
    int tn = wave % nt2;
    if (tm >= Mtiles2) return;         // wave-uniform: EXEC stays full
    int r    = lane & 15;
    int half = lane >> 4;

    const _Float16* a0 = A  + (size_t)(tm * 32 + r) * K + half * 8;
    const _Float16* a1 = a0 + (size_t)16 * K;
    const _Float16* b0 = Bt + (size_t)(tn * 32 + r) * K + half * 8;
    const _Float16* b1 = b0 + (size_t)16 * K;

    v8f c00 = {}, c01 = {}, c10 = {}, c11 = {};
    for (int k0 = 0; k0 < K; k0 += 32) {
        V16U A0, A1, B0, B1;
        A0.h[0] = *(const v8h*)(a0 + k0);
        A0.h[1] = *(const v8h*)(a0 + k0 + 16);
        A1.h[0] = *(const v8h*)(a1 + k0);
        A1.h[1] = *(const v8h*)(a1 + k0 + 16);
        B0.h[0] = *(const v8h*)(b0 + k0);
        B0.h[1] = *(const v8h*)(b0 + k0 + 16);
        B1.h[0] = *(const v8h*)(b1 + k0);
        B1.h[1] = *(const v8h*)(b1 + k0 + 16);
        c00 = WMMA16(A0.v, B0.v, c00);
        c01 = WMMA16(A0.v, B1.v, c01);
        c10 = WMMA16(A1.v, B0.v, c10);
        c11 = WMMA16(A1.v, B1.v, c11);
    }
    // C/D layout: VGPR j -> M = tileM + half*8 + j, N = tileN + r
    {
        float* p = C + (size_t)(tm * 32 + half * 8) * Nc + tn * 32 + r;
#pragma unroll
        for (int j = 0; j < 8; ++j) { p[(size_t)j * Nc] = c00[j]; p[(size_t)j * Nc + 16] = c01[j]; }
        p += (size_t)16 * Nc;
#pragma unroll
        for (int j = 0; j < 8; ++j) { p[(size_t)j * Nc] = c10[j]; p[(size_t)j * Nc + 16] = c11[j]; }
    }
}

// ---------------- per-node attention coefficients ----------------
// ai[n,h] = sum_c h[n,h,c]*att[h, 0:C],  aj[n,h] = sum_c h[n,h,c]*att[h, C:2C]
__global__ void node_coef(const float* __restrict__ h,
                          const float* __restrict__ att,
                          float* __restrict__ ai, float* __restrict__ aj,
                          int nodes, int H, int Cc) {
    int idx = blockIdx.x * blockDim.x + threadIdx.x;
    if (idx >= nodes * H) return;
    int node = idx / H, head = idx % H;
    const float* hv = h + (size_t)node * H * Cc + head * Cc;
    const float* ad = att + head * 2 * Cc;
    const float* as = ad + Cc;
    float si = 0.f, sj = 0.f;
    for (int c = 0; c < Cc; ++c) { si += hv[c] * ad[c]; sj += hv[c] * as[c]; }
    ai[idx] = si;
    aj[idx] = sj;
}

// ---------------- edge pass 1: leaky-relu alpha + segment max ----------------
__global__ void edge_max(const int* __restrict__ ei,
                         const float* __restrict__ ai, const float* __restrict__ aj,
                         float* __restrict__ alpha, unsigned* __restrict__ mmax,
                         int H) {
    int idx = blockIdx.x * blockDim.x + threadIdx.x;
    if (idx >= ETOT * H) return;
    int e = idx / H, head = idx % H;
    int s_, d_;
    if (e < E0C) { s_ = ei[e]; d_ = ei[E0C + e]; } else { s_ = d_ = e - E0C; }
    float a = ai[(size_t)d_ * H + head] + aj[(size_t)s_ * H + head];
    a = (a >= 0.f) ? a : NEG_SLOPE * a;
    alpha[idx] = a;
    atomicMax(mmax + (size_t)d_ * H + head, fenc(a));
}

// ---------------- edge pass 2: exp(alpha - max) + segment sum ----------------
__global__ void edge_expsum(const int* __restrict__ ei,
                            float* __restrict__ alpha,
                            const unsigned* __restrict__ mmax,
                            float* __restrict__ ssum, int H) {
    int idx = blockIdx.x * blockDim.x + threadIdx.x;
    if (idx >= ETOT * H) return;
    int e = idx / H, head = idx % H;
    int d_;
    if (e < E0C) { d_ = ei[E0C + e]; } else { d_ = e - E0C; }
    float m  = fdec(mmax[(size_t)d_ * H + head]);
    float ex = __expf(alpha[idx] - m);
    alpha[idx] = ex;                 // reuse as exp weight storage
    atomicAdd(ssum + (size_t)d_ * H + head, ex);
}

// ---------------- edge pass 3: weighted scatter-add ----------------
// one wave per edge; lane -> contiguous channel, coalesced 128B gather + atomics
__global__ void edge_aggregate(const int* __restrict__ ei,
                               const float* __restrict__ expw,
                               const float* __restrict__ ssum,
                               const float* __restrict__ h,
                               float* __restrict__ agg, int H, int Cc) {
    int gw   = (int)((blockIdx.x * blockDim.x + threadIdx.x) >> 5);
    int lane = threadIdx.x & 31;
    if (gw >= ETOT) return;
    int s_, d_;
    if (gw < E0C) { s_ = ei[gw]; d_ = ei[E0C + gw]; } else { s_ = d_ = gw - E0C; }
    int HC = H * Cc;
    const float* hs = h   + (size_t)s_ * HC;
    float*       ad = agg + (size_t)d_ * HC;
    for (int i = 0; i < HC; i += 32) {
        int ch   = i + lane;
        int head = ch / Cc;
        float w  = expw[(size_t)gw * H + head] / ssum[(size_t)d_ * H + head];
        atomicAdd(ad + ch, w * hs[ch]);
    }
}

// ---------------- fused bias + ELU + f16 cast (layer-1 output) ----------------
__global__ void elu_bias_cvt(const float* __restrict__ agg,
                             const float* __restrict__ bias,
                             _Float16* __restrict__ out, int total, int HC) {
    int i = blockIdx.x * blockDim.x + threadIdx.x;
    if (i >= total) return;
    float v = agg[i] + bias[i % HC];
    v = (v > 0.f) ? v : (__expf(v) - 1.f);
    out[i] = (_Float16)v;
}

// ---------------- final bias add ----------------
__global__ void add_bias(float* __restrict__ out, const float* __restrict__ bias,
                         int total, int Cc) {
    int i = blockIdx.x * blockDim.x + threadIdx.x;
    if (i < total) out[i] += bias[i % Cc];
}

// =====================================================================
extern "C" void kernel_launch(void* const* d_in, const int* in_sizes, int n_in,
                              void* d_out, int out_size, void* d_ws, size_t ws_size,
                              hipStream_t stream) {
    (void)in_sizes; (void)n_in; (void)out_size; (void)ws_size;

    const float* x    = (const float*)d_in[0];
    const int*   ei   = (const int*)  d_in[1];
    const float* W1   = (const float*)d_in[2];
    const float* att1 = (const float*)d_in[3];
    const float* b1   = (const float*)d_in[4];
    const float* W2   = (const float*)d_in[5];
    const float* att2 = (const float*)d_in[6];
    const float* b2   = (const float*)d_in[7];
    float* out = (float*)d_out;

    // -------- workspace arena --------
    char* base = (char*)d_ws;
    size_t off = 0;
    auto carve = [&](size_t bytes) -> char* {
        char* p = base + off;
        off = (off + bytes + 255) & ~(size_t)255;
        return p;
    };
    // padded to MPAD rows where the buffer feeds / is fed by the WMMA GEMM
    _Float16* XH    = (_Float16*)carve((size_t)MPAD * INDIM * 2);   // x f16; reused as layer-2 act f16
    _Float16* W1T   = (_Float16*)carve((size_t)INDIM * HIDDIM * 2);
    _Float16* W2T   = (_Float16*)carve((size_t)HIDDIM * OUTDIM * 2);
    float*    HBUF  = (float*)   carve((size_t)MPAD * HIDDIM * 4);  // h1; reused as h2
    float*    AGG1  = (float*)   carve((size_t)NN * HIDDIM * 4);
    float*    AI    = (float*)   carve((size_t)NN * H1C * 4);
    float*    AJ    = (float*)   carve((size_t)NN * H1C * 4);
    unsigned* MM    = (unsigned*)carve((size_t)NN * H1C * 4);
    float*    SS    = (float*)   carve((size_t)NN * H1C * 4);
    float*    ALPHA = (float*)   carve((size_t)ETOT * H1C * 4);

    const int TB = 256;
    auto nb = [](long long n, int tb) { return (unsigned)((n + tb - 1) / tb); };
    const int MT2 = MPAD / 32;         // 1563 row-tiles of 32

    // ================= layer 1 =================
    cvt_f32_to_f16<<<nb((long long)NN * INDIM, TB), TB, 0, stream>>>(x, XH, NN * INDIM);
    xpose_cvt_f16 <<<nb((long long)INDIM * HIDDIM, TB), TB, 0, stream>>>(W1, W1T, INDIM, HIDDIM);
    xpose_cvt_f16 <<<nb((long long)HIDDIM * OUTDIM, TB), TB, 0, stream>>>(W2, W2T, HIDDIM, OUTDIM);

    // h1 = x @ W1   (WMMA f16->f32, 2x2 wave tiles)
    {
        long long waves = (long long)MT2 * (HIDDIM / 32);
        gemm_f16_wmma<<<nb(waves * 32, 128), 128, 0, stream>>>(XH, W1T, HBUF, MT2, HIDDIM, INDIM);
    }

    node_coef<<<nb((long long)NN * H1C, TB), TB, 0, stream>>>(HBUF, att1, AI, AJ, NN, H1C, C1C);

    hipMemsetAsync(MM,   0, (size_t)NN * H1C * 4, stream);
    hipMemsetAsync(SS,   0, (size_t)NN * H1C * 4, stream);
    hipMemsetAsync(AGG1, 0, (size_t)NN * HIDDIM * 4, stream);

    edge_max   <<<nb((long long)ETOT * H1C, TB), TB, 0, stream>>>(ei, AI, AJ, ALPHA, MM, H1C);
    edge_expsum<<<nb((long long)ETOT * H1C, TB), TB, 0, stream>>>(ei, ALPHA, MM, SS, H1C);
    edge_aggregate<<<nb((long long)ETOT * 32, TB), TB, 0, stream>>>(ei, ALPHA, SS, HBUF, AGG1, H1C, C1C);

    // h1e = elu(agg + b1) -> f16 (reuse XH; pad rows keep garbage, never read back)
    elu_bias_cvt<<<nb((long long)NN * HIDDIM, TB), TB, 0, stream>>>(AGG1, b1, XH, NN * HIDDIM, HIDDIM);

    // ================= layer 2 =================
    // h2 = h1e @ W2 (reuse HBUF as [MPAD,64])
    {
        long long waves = (long long)MT2 * (OUTDIM / 32);
        gemm_f16_wmma<<<nb(waves * 32, 128), 128, 0, stream>>>(XH, W2T, HBUF, MT2, OUTDIM, HIDDIM);
    }

    node_coef<<<nb((long long)NN * H2C, TB), TB, 0, stream>>>(HBUF, att2, AI, AJ, NN, H2C, C2C);

    hipMemsetAsync(MM,  0, (size_t)NN * H2C * 4, stream);
    hipMemsetAsync(SS,  0, (size_t)NN * H2C * 4, stream);
    hipMemsetAsync(out, 0, (size_t)NN * OUTDIM * 4, stream);

    edge_max   <<<nb((long long)ETOT * H2C, TB), TB, 0, stream>>>(ei, AI, AJ, ALPHA, MM, H2C);
    edge_expsum<<<nb((long long)ETOT * H2C, TB), TB, 0, stream>>>(ei, ALPHA, MM, SS, H2C);
    edge_aggregate<<<nb((long long)ETOT * 32, TB), TB, 0, stream>>>(ei, ALPHA, SS, HBUF, out, H2C, C2C);

    add_bias<<<nb((long long)NN * OUTDIM, TB), TB, 0, stream>>>(out, b2, NN * OUTDIM, OUTDIM);
}
